// DenseCRF_84533546320361
// MI455X (gfx1250) — matrix operand order repaired and checked by reading, not compiled, not run on hardware
//
#include <hip/hip_runtime.h>
#include <cstdint>

// ---------------- Problem constants ----------------
#define NPIX   9216            // 96*96
#define WID    96
#define NCLS   21
#define CPAD   32              // classes padded to 2x 16-wide WMMA tiles
#define KTILES (NPIX / 32)     // 288 k-tiles of 32 pixels
#define MTILES (NPIX / 16)     // 576 m-tiles of 16 rows
#define KPARTS 4               // K-dimension split (deterministic partial buffers)
#define PTILES (KTILES / KPARTS)  // 72 k-tiles per part
#define ITERS  5

typedef __attribute__((ext_vector_type(16))) __bf16 v16bf;
typedef __attribute__((ext_vector_type(8)))  __bf16 v8bf;
typedef __attribute__((ext_vector_type(8)))  float  v8f;

union AB16 { v16bf v; v8bf h[2]; };

// Async global->LDS copy path (CDNA5), guarded so we always compile.
#if defined(__has_builtin)
#if __has_builtin(__builtin_amdgcn_global_load_async_to_lds_b128)
#define USE_ASYNC_LDS 1
#endif
#if __has_builtin(__builtin_amdgcn_s_wait_asynccnt)
#define HAVE_WAIT_ASYNC 1
#endif
#endif

#if defined(USE_ASYNC_LDS)
// Builtin signature (probe-verified): param0 = int4* in AS(1), param1 = int4* in AS(3),
// then immediate offset + cpol.
typedef int v4i_t __attribute__((vector_size(16)));
typedef __attribute__((address_space(1))) v4i_t* as1_v4i;
typedef __attribute__((address_space(3))) v4i_t* as3_v4i;
#endif

// B-fragment swizzle for V_WMMA_*_16X16X32 (wave32), in this code's contiguous k order (the A loads match):
// B is KxN (32 x 16): lanes 0-15 hold K=0..15 (elem = k), lanes 16-31 hold K=16..31
// (elem = k-16); column n = lane % 16. Two class-tiles ct=0 (c 0..15), ct=1 (c 16..31).
__device__ __forceinline__ size_t swz_idx(int p, int c) {
    int t    = p >> 5;          // k-tile (32 pixels)
    int kin  = p & 31;
    int half = kin >> 4;        // lane group
    int e    = kin & 15;        // element within lane's 16 bf16
    int lane = half * 16 + (c & 15);
    int ct   = c >> 4;
    return ((((size_t)ct * KTILES + t) * 32 + lane) << 4) + e;
}

// ------------- Phase A: build combined Gaussian kernel matrix (bf16) -------------
// K_comb[i][j] = 10*exp(-0.5*||f_bi_i - f_bi_j||^2) + 3*exp(-0.5*||f_sm_i - f_sm_j||^2)
__global__ void k_build(const float* __restrict__ x, __bf16* __restrict__ Kc) {
    const int j = blockIdx.x * blockDim.x + threadIdx.x;   // 0..NPIX-1 (exact)
    const int i = blockIdx.y;

    const int iy = i / WID, ix = i - iy * WID;
    const int jy = j / WID, jx = j - jy * WID;

    const float ir = x[i], ig = x[NPIX + i], ib = x[2 * NPIX + i];
    const float jr = x[j], jg = x[NPIX + j], jb = x[2 * NPIX + j];

    const float dy = (float)(iy - jy);
    const float dx = (float)(ix - jx);
    const float dr = ir - jr, dg = ig - jg, db = ib - jb;

    const float pos2 = dy * dy + dx * dx;
    const float col2 = dr * dr + dg * dg + db * db;

    // 0.5/std^2 factors: prox 30 -> 1/1800, color 0.1 -> 50, smooth 3 -> 1/18
    const float dbi = pos2 * (1.0f / 1800.0f) + col2 * 50.0f;
    const float dsm = pos2 * (1.0f / 18.0f);

    const float val = 10.0f * __expf(-dbi) + 3.0f * __expf(-dsm);
    Kc[(size_t)i * NPIX + j] = (__bf16)val;
}

// ------------- init: unary = yhat (padded), out = softmax(yhat) into swizzled B -------------
__global__ void k_init(const float* __restrict__ yhat, float* __restrict__ unary,
                       __bf16* __restrict__ outTsw) {
    const int p = blockIdx.x * blockDim.x + threadIdx.x;

    float u[NCLS];
    float m = -3.0e38f;
#pragma unroll
    for (int c = 0; c < NCLS; ++c) { u[c] = yhat[(size_t)c * NPIX + p]; m = fmaxf(m, u[c]); }
    float e[NCLS], s = 0.0f;
#pragma unroll
    for (int c = 0; c < NCLS; ++c) { e[c] = __expf(u[c] - m); s += e[c]; }
    const float inv = 1.0f / s;

#pragma unroll
    for (int c = 0; c < CPAD; ++c) {
        unary[(size_t)p * CPAD + c] = (c < NCLS) ? u[c] : 0.0f;
        const float pr = (c < NCLS) ? e[c] * inv : 0.0f;
        outTsw[swz_idx(p, c)] = (__bf16)pr;
    }
}

// ------------- GEMM: pw_part = K_comb[mtile rows, kpart] @ outT  (bf16 WMMA) -------------
// Block = 8 waves, all sharing one K-part. B slice for the part (144 KB) is staged
// once into LDS via async global->LDS DMA, then the k-loop reads B via DS.
__global__ void __launch_bounds__(256) k_gemm(const __bf16* __restrict__ Kc,
                                              const __bf16* __restrict__ outTsw,
                                              float* __restrict__ pwParts) {
    extern __shared__ __bf16 ldsB[];               // 2 * PTILES * 512 bf16 = 147456 B

    const int lane = threadIdx.x;                  // 0..31 (wave32)
    const int part = blockIdx.x / (MTILES / 8);    // 72 blocks per part -> block-uniform
    const int mblk = blockIdx.x - part * (MTILES / 8);
    const int m0   = (mblk * 8 + threadIdx.y) * 16;

    const int r15 = lane & 15;
    const int sel = lane >> 4;
    const int tid = threadIdx.y * 32 + lane;
    const int t0  = part * PTILES;                 // first k-tile of this part

    // ---- Stage B slice into LDS (2 class-tiles x PTILES k-tiles x 1KB) ----
#pragma unroll
    for (int ct = 0; ct < 2; ++ct) {
        const __bf16* gsrc = outTsw + ((size_t)ct * KTILES + t0) * 512;
        __bf16* ldst = ldsB + ct * (PTILES * 512);
#pragma unroll
        for (int s = 0; s < (PTILES * 512) / (256 * 8); ++s) {   // 18 x 16B per thread
            const int idx = (s * 256 + tid) * 8;                 // bf16 element index
#if defined(USE_ASYNC_LDS)
            __builtin_amdgcn_global_load_async_to_lds_b128(
                (as1_v4i)(uintptr_t)(gsrc + idx),
                (as3_v4i)(uint32_t)(uintptr_t)(ldst + idx),
                0, 0);
#else
            *(v8bf*)(ldst + idx) = *(const v8bf*)(gsrc + idx);
#endif
        }
    }
#if defined(USE_ASYNC_LDS)
#if defined(HAVE_WAIT_ASYNC)
    __builtin_amdgcn_s_wait_asynccnt(0);
#else
    asm volatile("s_wait_asynccnt 0x0" ::: "memory");
#endif
#endif
    __syncthreads();

    // A fragment: lane<16 -> row m0+lane, K = 0..15; lane>=16 -> K = 16..31
    const __bf16* arow = Kc + (size_t)(m0 + r15) * NPIX + sel * 16;
    const __bf16* lb   = ldsB + (size_t)lane * 16;

    v8f acc0 = {};
    v8f acc1 = {};

    const int kbeg = part * (NPIX / KPARTS);       // 72 steps of 32, 12 x unroll-6

#pragma unroll 6
    for (int kof = 0; kof < NPIX / KPARTS; kof += 32) {
        AB16 a, b0, b1;
        const v8bf* pa = (const v8bf*)(arow + kbeg + kof);
        a.h[0] = pa[0];                            // K = kk + sel*16 + 0..7
        a.h[1] = pa[1];                            // K = kk + sel*16 + 8..15

        // prefetch A two k-steps ahead (speculative global_prefetch)
        __builtin_prefetch((const void*)(arow + kbeg + kof + 64), 0, 1);

        const __bf16* lt = lb + (size_t)(kof >> 5) * 512;       // local k-tile
        const v8bf* pb0 = (const v8bf*)lt;
        const v8bf* pb1 = (const v8bf*)(lt + (size_t)PTILES * 512);
        b0.h[0] = pb0[0]; b0.h[1] = pb0[1];
        b1.h[0] = pb1[0]; b1.h[1] = pb1[1];

        acc0 = __builtin_amdgcn_wmma_f32_16x16x32_bf16(false, a.v, false, b0.v,
                                                       (short)0, acc0, false, false);
        acc1 = __builtin_amdgcn_wmma_f32_16x16x32_bf16(false, a.v, false, b1.v,
                                                       (short)0, acc1, false, false);
    }

    // D layout: VGPR r -> row m = r + 8*sel, col n = lane & 15
    float* pwp = pwParts + (size_t)part * NPIX * CPAD;
#pragma unroll
    for (int r = 0; r < 8; ++r) {
        const int m = r + 8 * sel;
        pwp[(size_t)(m0 + m) * CPAD + r15]      = acc0[r];
        pwp[(size_t)(m0 + m) * CPAD + 16 + r15] = acc1[r];
    }
}

// ------------- update: sum K-partials, Potts transform, unary += pw, softmax, re-swizzle -------------
__global__ void k_update(const float* __restrict__ pwParts, float* __restrict__ unary,
                         __bf16* __restrict__ outTsw, float* __restrict__ outp) {
    const int p = blockIdx.x * blockDim.x + threadIdx.x;

    float pw[NCLS];
    float s = 0.0f;
#pragma unroll
    for (int c = 0; c < NCLS; ++c) {
        float v = 0.0f;
#pragma unroll
        for (int q = 0; q < KPARTS; ++q)
            v += pwParts[(size_t)q * NPIX * CPAD + (size_t)p * CPAD + c];
        pw[c] = v;
        s += v;
    }

    float u[NCLS];
    float m = -3.0e38f;
#pragma unroll
    for (int c = 0; c < NCLS; ++c) {
        // (pw @ (ones - eye))[c] = rowsum - pw[c]
        float nu = unary[(size_t)p * CPAD + c] + (s - pw[c]);
        unary[(size_t)p * CPAD + c] = nu;
        u[c] = nu;
        m = fmaxf(m, nu);
    }
    float e[NCLS], es = 0.0f;
#pragma unroll
    for (int c = 0; c < NCLS; ++c) { e[c] = __expf(u[c] - m); es += e[c]; }
    const float inv = 1.0f / es;

#pragma unroll
    for (int c = 0; c < CPAD; ++c) {
        const float pr = (c < NCLS) ? e[c] * inv : 0.0f;
        outTsw[swz_idx(p, c)] = (__bf16)pr;
        if (c < NCLS) outp[(size_t)c * NPIX + p] = pr;   // (C,H,W) fp32; last iter wins
    }
}

// ---------------- host-side launcher ----------------
extern "C" void kernel_launch(void* const* d_in, const int* in_sizes, int n_in,
                              void* d_out, int out_size, void* d_ws, size_t ws_size,
                              hipStream_t stream) {
    (void)in_sizes; (void)n_in; (void)out_size; (void)ws_size;

    const float* x    = (const float*)d_in[0];   // (3, 96, 96)
    const float* yhat = (const float*)d_in[1];   // (21, 96, 96)
    // d_in[2] = mu (Potts, ones - eye) is hardcoded algebraically.
    float* outp = (float*)d_out;                 // (21, 96, 96) fp32

    // Workspace carve-out (256B aligned):
    //   Kc      : NPIX*NPIX bf16   (~162 MB)
    //   unary   : NPIX*CPAD f32
    //   outTsw  : NPIX*CPAD bf16   (WMMA B-fragment swizzled)
    //   pwParts : KPARTS*NPIX*CPAD f32
    char* ws = (char*)d_ws;
    size_t off = 0;
    __bf16* Kc = (__bf16*)(ws + off);
    off += ((size_t)NPIX * NPIX * sizeof(__bf16) + 255) & ~(size_t)255;
    float* unary = (float*)(ws + off);
    off += ((size_t)NPIX * CPAD * sizeof(float) + 255) & ~(size_t)255;
    __bf16* outTsw = (__bf16*)(ws + off);
    off += ((size_t)NPIX * CPAD * sizeof(__bf16) + 255) & ~(size_t)255;
    float* pwParts = (float*)(ws + off);

    const size_t ldsBytes = (size_t)2 * PTILES * 512 * sizeof(__bf16);   // 147456 B

    // Phase A: combined Gaussian kernel matrix (one-time, ~85M exp-pairs)
    k_build<<<dim3(NPIX / 256, NPIX), 256, 0, stream>>>(x, Kc);

    // Init: unary + softmax + swizzled B operand
    k_init<<<NPIX / 256, 256, 0, stream>>>(yhat, unary, outTsw);

    // Mean-field iterations: WMMA message passing + compatibility/softmax update
    for (int it = 0; it < ITERS; ++it) {
        k_gemm<<<(MTILES * KPARTS) / 8, dim3(32, 8), ldsBytes, stream>>>(Kc, outTsw, pwParts);
        k_update<<<NPIX / 256, 256, 0, stream>>>(pwParts, unary, outTsw, outp);
    }
}
